// MultiHeadAttention_58179626991887
// MI455X (gfx1250) — compile-verified, matrix operand order checked by reading, not run on hardware
//
#include <hip/hip_runtime.h>

// ---------------------------------------------------------------------------
// Causal MHA forward for B=4, S=2048, D=1024, H=16, DH=64 on gfx1250 (MI455X).
// All matmuls via v_wmma_f32_16x16x32_bf16 (wave32). Intermediates in bf16.
// GEMM waves compute 32x64 tiles (2 A-frags x 4 B-frags -> 8 WMMA/k-step) for
// ~22 FLOP/byte of cache traffic; WGP-scope global_prefetch_b8 (locality 3)
// pulls upcoming tiles into the near caches, not just GL2.
// ---------------------------------------------------------------------------

typedef __attribute__((ext_vector_type(4)))  __bf16 v4bf;
typedef __attribute__((ext_vector_type(8)))  __bf16 v8bf;
typedef __attribute__((ext_vector_type(16))) __bf16 v16bf;
typedef __attribute__((ext_vector_type(4)))  float  v4f;
typedef __attribute__((ext_vector_type(8)))  float  v8f;

#define NB   4
#define NS   2048
#define ND   1024
#define NH   16
#define NDH  64
#define NM   (NB * NS)          // 8192 flattened rows
#define ROWT (NM / 32)          // 256 row tiles of 32
#define COLG (ND / 64)          // 16 column groups of 64

__device__ __forceinline__ v16bf cat16(v8bf lo, v8bf hi) {
  return __builtin_shufflevector(lo, hi, 0,1,2,3,4,5,6,7,8,9,10,11,12,13,14,15);
}
__device__ __forceinline__ v8bf ldg8(const __bf16* p) {
  return *reinterpret_cast<const v8bf*>(p);
}

// ---- kernel 0: fp32 -> bf16 cast of X (vectorized x4) ----------------------
__global__ void k_convert_x(const float* __restrict__ X, __bf16* __restrict__ Xb, int n4) {
  int t = blockIdx.x * blockDim.x + threadIdx.x;
  if (t >= n4) return;
  v4f x = *reinterpret_cast<const v4f*>(X + 4 * (size_t)t);
  v4bf y;
  y.x = (__bf16)x.x; y.y = (__bf16)x.y; y.z = (__bf16)x.z; y.w = (__bf16)x.w;
  *reinterpret_cast<v4bf*>(Xb + 4 * (size_t)t) = y;
}

// ---- kernel 1: transpose + cast the 4 weight matrices to bf16 [n][k] -------
__global__ void k_transpose_w(const float* __restrict__ Wq, const float* __restrict__ Wk,
                              const float* __restrict__ Wv, const float* __restrict__ Wo,
                              __bf16* __restrict__ Wt) {
  int t = blockIdx.x * blockDim.x + threadIdx.x;   // 4 * 1024 * 1024 threads
  int mat = t >> 20;
  int idx = t & 0xFFFFF;
  int k = idx >> 10;
  int n = idx & 1023;
  const float* W = (mat == 0) ? Wq : (mat == 1) ? Wk : (mat == 2) ? Wv : Wo;
  Wt[((size_t)mat << 20) + ((size_t)n << 10) + k] = (__bf16)W[((size_t)k << 10) + n];
}

// ---- kernel 2: fused QKV projection; Q,K -> [B,H,S,DH], V -> [B,H,DH,S] ----
// One wave computes a 32x64 output tile: 2 A-fragments x 4 B-fragments,
// 8 accumulators, 8 WMMAs per 32-wide k-step.
__global__ void k_gemm_qkv(const __bf16* __restrict__ Xb, const __bf16* __restrict__ Wt,
                           __bf16* __restrict__ Qb, __bf16* __restrict__ Kb,
                           __bf16* __restrict__ Vtb) {
  const int lane = threadIdx.x & 31;
  const int wid  = (blockIdx.x * blockDim.x + threadIdx.x) >> 5;
  const int mat  = wid / (ROWT * COLG);       // 0=Q,1=K,2=V
  const int rem  = wid % (ROWT * COLG);
  const int rt   = rem >> 4;                  // 32-row tile
  const int cg   = rem & 15;                  // 64-wide column group
  const int l16  = lane & 15, half = lane >> 4;

  const __bf16* wmat  = Wt + ((size_t)mat << 20);
  const __bf16* arow0 = Xb + (size_t)(rt * 32 + l16) * ND;      // A: lane = row m
  const __bf16* arow1 = arow0 + (size_t)16 * ND;
  v8f acc[2][4] = {};
  for (int kk = 0; kk < ND; kk += 32) {
    v16bf a0 = cat16(ldg8(arow0 + kk + half * 8), ldg8(arow0 + kk + 16 + half * 8));
    v16bf a1 = cat16(ldg8(arow1 + kk + half * 8), ldg8(arow1 + kk + 16 + half * 8));
    __builtin_prefetch(arow0 + kk + 256, 0, 3);   // WGP-scope prefetch, 8 steps ahead
    __builtin_prefetch(arow1 + kk + 256, 0, 3);
#pragma unroll
    for (int c = 0; c < 4; ++c) {
      const __bf16* brow = wmat + (size_t)(cg * 64 + c * 16 + l16) * ND; // B: lane = col n
      v16bf b = cat16(ldg8(brow + kk + half * 8), ldg8(brow + kk + 16 + half * 8));
      __builtin_prefetch(brow + kk + 256, 0, 3);
      acc[0][c] = __builtin_amdgcn_wmma_f32_16x16x32_bf16(false, a0, false, b,
                                                          (short)0, acc[0][c], false, false);
      acc[1][c] = __builtin_amdgcn_wmma_f32_16x16x32_bf16(false, a1, false, b,
                                                          (short)0, acc[1][c], false, false);
    }
  }
#pragma unroll
  for (int mt = 0; mt < 2; ++mt) {
#pragma unroll
    for (int c = 0; c < 4; ++c) {
      int outcol = cg * 64 + c * 16 + l16;
      int h = outcol >> 6, dh = outcol & 63;
#pragma unroll
      for (int r = 0; r < 8; ++r) {
        int sflat = rt * 32 + mt * 16 + half * 8 + r;
        int bb = sflat >> 11, s = sflat & 2047;
        __bf16 v = (__bf16)acc[mt][c][r];
        if (mat == 0)      Qb [(((size_t)(bb * NH + h) * NS + s) << 6) + dh] = v;
        else if (mat == 1) Kb [(((size_t)(bb * NH + h) * NS + s) << 6) + dh] = v;
        else               Vtb[(((size_t)(bb * NH + h) * NDH + dh) << 11) + s] = v;
      }
    }
  }
}

// ---- kernel 3: causal flash attention, one wave per (b,h, 16-query tile) ---
__global__ void k_attn(const __bf16* __restrict__ Qb, const __bf16* __restrict__ Kb,
                       const __bf16* __restrict__ Vtb, __bf16* __restrict__ Ctx) {
  __shared__ __bf16 plds[8 * 16 * 32];   // 8 waves/block x (16x32 bf16 P tile)
  const int lane = threadIdx.x & 31;
  const int wib  = threadIdx.x >> 5;
  const int wid  = (blockIdx.x * blockDim.x + threadIdx.x) >> 5;
  const int bh   = wid >> 7;             // 0..63  (b*16+h)
  const int qt   = wid & 127;            // query tile
  const int l16  = lane & 15, half = lane >> 4;
  const int q0   = qt * 16;

  const __bf16* qbase = Qb  + (size_t)bh * NS * NDH;
  const __bf16* kbase = Kb  + (size_t)bh * NS * NDH;
  const __bf16* vbase = Vtb + (size_t)bh * NDH * NS;

  // Q A-fragments for the two k-dim steps (DH = 64 = 2 x 32), reused all loop
  const __bf16* qrow = qbase + (size_t)(q0 + l16) * NDH;
  v16bf aQ[2];
#pragma unroll
  for (int ks = 0; ks < 2; ++ks)
    aQ[ks] = cat16(ldg8(qrow + ks * 32 + half * 8),
                   ldg8(qrow + ks * 32 + 16 + half * 8));

  float mrow[8], lrow[8];
#pragma unroll
  for (int r = 0; r < 8; ++r) { mrow[r] = -__builtin_inff(); lrow[r] = 0.f; }
  v8f O[4] = {};

  const float scale = 0.03125f;                 // 1/sqrt(D) = 1/32 (matches ref)
  const float LOG2E = 1.4426950408889634f;
  __bf16* pbuf = plds + wib * (16 * 32);

  const int nsteps = (q0 + 47) >> 5;            // 32-key steps covering keys <= q0+15
  for (int st = 0; st < nsteps; ++st) {
    const int kb = st << 5;
    // WGP-scope prefetch of next key-block's K rows and V columns
    __builtin_prefetch(kbase + (size_t)(kb + 32 + l16) * NDH, 0, 3);
    __builtin_prefetch(kbase + (size_t)(kb + 48 + l16) * NDH, 0, 3);
    __builtin_prefetch(vbase + (size_t)l16 * NS + kb + 32, 0, 3);
    __builtin_prefetch(vbase + (size_t)(48 + l16) * NS + kb + 32, 0, 3);
    v8f sacc[2] = {};
#pragma unroll
    for (int t = 0; t < 2; ++t) {
      const __bf16* krow = kbase + (size_t)(kb + t * 16 + l16) * NDH;  // B: lane = key
#pragma unroll
      for (int ks = 0; ks < 2; ++ks) {
        v16bf bK = cat16(ldg8(krow + ks * 32 + half * 8),
                         ldg8(krow + ks * 32 + 16 + half * 8));
        sacc[t] = __builtin_amdgcn_wmma_f32_16x16x32_bf16(false, aQ[ks], false, bK,
                                                          (short)0, sacc[t], false, false);
      }
    }
    // online softmax (row stats live per accumulator slot r; 16-lane reductions)
    float p0[8], p1[8];
#pragma unroll
    for (int r = 0; r < 8; ++r) {
      int qg = q0 + half * 8 + r;
      float s0 = sacc[0][r] * scale;
      float s1 = sacc[1][r] * scale;
      if (kb + l16 > qg)      s0 = -__builtin_inff();   // causal mask
      if (kb + 16 + l16 > qg) s1 = -__builtin_inff();
      float rm = fmaxf(s0, s1);
#pragma unroll
      for (int d = 1; d < 16; d <<= 1) rm = fmaxf(rm, __shfl_xor(rm, d, 32));
      float mnew  = fmaxf(mrow[r], rm);
      float alpha = exp2f((mrow[r] - mnew) * LOG2E);
      float e0 = exp2f((s0 - mnew) * LOG2E);
      float e1 = exp2f((s1 - mnew) * LOG2E);
      float ps = e0 + e1;
#pragma unroll
      for (int d = 1; d < 16; d <<= 1) ps += __shfl_xor(ps, d, 32);
      lrow[r] = lrow[r] * alpha + ps;
      mrow[r] = mnew;
      p0[r] = e0; p1[r] = e1;
#pragma unroll
      for (int c = 0; c < 4; ++c) O[c][r] *= alpha;
    }
    // stage P (C/D layout) -> LDS row-major [q][key] to re-load as an A-fragment
#pragma unroll
    for (int r = 0; r < 8; ++r) {
      int row = half * 8 + r;
      pbuf[row * 32 + l16]      = (__bf16)p0[r];
      pbuf[row * 32 + 16 + l16] = (__bf16)p1[r];
    }
    v16bf aP = cat16(*reinterpret_cast<const v8bf*>(pbuf + l16 * 32 + half * 8),
                     *reinterpret_cast<const v8bf*>(pbuf + l16 * 32 + 16 + half * 8));
#pragma unroll
    for (int c = 0; c < 4; ++c) {
      const __bf16* vrow = vbase + (size_t)(c * 16 + l16) * NS + kb;   // B: lane = dh
      v16bf bV = cat16(ldg8(vrow + half * 8), ldg8(vrow + 16 + half * 8));
      O[c] = __builtin_amdgcn_wmma_f32_16x16x32_bf16(false, aP, false, bV,
                                                     (short)0, O[c], false, false);
    }
  }
  // finalize and store ctx as bf16 in [B, S, D] layout for the output GEMM
  const int bb = bh >> 4, h = bh & 15;
#pragma unroll
  for (int r = 0; r < 8; ++r) {
    float inv = 1.f / lrow[r];
    int s = q0 + half * 8 + r;
    size_t base = ((size_t)(bb * NS + s)) * ND + h * NDH;
#pragma unroll
    for (int c = 0; c < 4; ++c)
      Ctx[base + c * 16 + l16] = (__bf16)(O[c][r] * inv);
  }
}

// ---- kernel 4: output projection ctx @ Wo + bo -> fp32 (32x64 wave tiles) --
__global__ void k_gemm_out(const __bf16* __restrict__ Ctxb, const __bf16* __restrict__ Wot,
                           const float* __restrict__ bo, float* __restrict__ Out) {
  const int lane = threadIdx.x & 31;
  const int wid  = (blockIdx.x * blockDim.x + threadIdx.x) >> 5;
  const int rt   = wid >> 4;
  const int cg   = wid & 15;
  const int l16  = lane & 15, half = lane >> 4;

  const __bf16* arow0 = Ctxb + (size_t)(rt * 32 + l16) * ND;
  const __bf16* arow1 = arow0 + (size_t)16 * ND;
  v8f acc[2][4] = {};
  for (int kk = 0; kk < ND; kk += 32) {
    v16bf a0 = cat16(ldg8(arow0 + kk + half * 8), ldg8(arow0 + kk + 16 + half * 8));
    v16bf a1 = cat16(ldg8(arow1 + kk + half * 8), ldg8(arow1 + kk + 16 + half * 8));
    if (kk < ND - 256) {                        // keep tail prefetch in-bounds
      __builtin_prefetch(arow0 + kk + 256, 0, 3);
      __builtin_prefetch(arow1 + kk + 256, 0, 3);
    }
#pragma unroll
    for (int c = 0; c < 4; ++c) {
      const __bf16* brow = Wot + (size_t)(cg * 64 + c * 16 + l16) * ND;
      v16bf b = cat16(ldg8(brow + kk + half * 8), ldg8(brow + kk + 16 + half * 8));
      __builtin_prefetch(brow + kk + 256, 0, 3);
      acc[0][c] = __builtin_amdgcn_wmma_f32_16x16x32_bf16(false, a0, false, b,
                                                          (short)0, acc[0][c], false, false);
      acc[1][c] = __builtin_amdgcn_wmma_f32_16x16x32_bf16(false, a1, false, b,
                                                          (short)0, acc[1][c], false, false);
    }
  }
#pragma unroll
  for (int mt = 0; mt < 2; ++mt) {
#pragma unroll
    for (int c = 0; c < 4; ++c) {
      int outcol = cg * 64 + c * 16 + l16;
      float bias = bo[outcol];
#pragma unroll
      for (int r = 0; r < 8; ++r) {
        int sflat = rt * 32 + mt * 16 + half * 8 + r;
        Out[(size_t)sflat * ND + outcol] = acc[mt][c][r] + bias;
      }
    }
  }
}

// ---------------------------------------------------------------------------
extern "C" void kernel_launch(void* const* d_in, const int* in_sizes, int n_in,
                              void* d_out, int out_size, void* d_ws, size_t ws_size,
                              hipStream_t stream) {
  (void)in_sizes; (void)n_in; (void)out_size; (void)ws_size;
  const float* X  = (const float*)d_in[0];
  const float* Wq = (const float*)d_in[1];
  const float* Wk = (const float*)d_in[2];
  const float* Wv = (const float*)d_in[3];
  const float* Wo = (const float*)d_in[4];
  const float* bo = (const float*)d_in[5];
  float* Out = (float*)d_out;

  // workspace layout (bf16 = 2 B):
  //   Xb   : 8192x1024            = 16 MiB
  //   Wt   : 4 x 1024x1024 (n,k)  =  8 MiB
  //   Qb   : [B,H,S,DH]           = 16 MiB
  //   Kb   : [B,H,S,DH]           = 16 MiB
  //   Vtb  : [B,H,DH,S]           = 16 MiB
  //   Ctxb : [B,S,D]              = 16 MiB      total 88 MiB
  char* ws = (char*)d_ws;
  __bf16* Xb   = (__bf16*)(ws);
  __bf16* Wt   = (__bf16*)(ws + (size_t)16 * 1024 * 1024);
  __bf16* Qb   = (__bf16*)(ws + (size_t)24 * 1024 * 1024);
  __bf16* Kb   = (__bf16*)(ws + (size_t)40 * 1024 * 1024);
  __bf16* Vtb  = (__bf16*)(ws + (size_t)56 * 1024 * 1024);
  __bf16* Ctxb = (__bf16*)(ws + (size_t)72 * 1024 * 1024);

  // 0) cast X -> bf16
  {
    int n4 = NM * ND / 4;                      // 2,097,152
    k_convert_x<<<n4 / 256, 256, 0, stream>>>(X, Xb, n4);
  }
  // 1) transpose + cast weights
  k_transpose_w<<<(4 * 1024 * 1024) / 256, 256, 0, stream>>>(Wq, Wk, Wv, Wo, Wt);
  // 2) fused QKV projection: 3 * 256 * 16 waves (32x64 tiles)
  k_gemm_qkv<<<(3 * ROWT * COLG * 32) / 256, 256, 0, stream>>>(Xb, Wt, Qb, Kb, Vtb);
  // 3) flash attention: B*H*(S/16) = 8192 waves
  k_attn<<<(NB * NH * (NS / 16) * 32) / 256, 256, 0, stream>>>(Qb, Kb, Vtb, Ctxb);
  // 4) output projection + bias: 256 * 16 waves (32x64 tiles)
  k_gemm_out<<<(ROWT * COLG * 32) / 256, 256, 0, stream>>>(Ctxb, Wt + ((size_t)3 << 20),
                                                           bo, Out);
}
